// TransformerConv_32487132627456
// MI455X (gfx1250) — matrix operand order repaired and checked by reading, not compiled, not run on hardware
//
#include <hip/hip_runtime.h>
#include <hip/hip_bf16.h>

// ---- problem constants (from reference) ----
#define NN   50000     // nodes
#define EE   800000    // edges
#define DIN  64
#define HH   4         // heads
#define CC   64        // channels per head
#define HC   256       // H*C
#define EDF  11        // edge feature dim
#define FUSED 1024     // q|k|v|skip fused columns

typedef __attribute__((ext_vector_type(16))) __bf16 v16bf;
typedef __attribute__((ext_vector_type(8)))  float  v8f;

// Packed-B geometry: fragment block = 32 lanes x 16 bf16 = 1024 bytes
#define WCAT_CT   64   // 1024 cols / 16
#define WCAT_KS   2    // K=64 / 32
#define WLIN_CT   4    // 64 cols / 16
#define WLIN_KS   8    // K=256 / 32

// ---------- ordered-uint encoding for float atomic max ----------
__device__ __forceinline__ unsigned encOrdered(float f) {
    unsigned u = __float_as_uint(f);
    return (u & 0x80000000u) ? ~u : (u | 0x80000000u);
}
__device__ __forceinline__ float decOrdered(unsigned u) {
    return (u & 0x80000000u) ? __uint_as_float(u & 0x7FFFFFFFu)
                             : __uint_as_float(~u);
}
#define ENC_NEG_INF 0x007FFFFFu   // encOrdered(-inf)

// ---------------------------------------------------------------
// Kernel 1: weight prep -> bf16 B fragments pre-packed in WMMA lane order.
// WcatP[(ct*2+ks)*32+lane][i] = Wfused[ks*32+(lane>>4)*16+i][ct*16+(lane&15)]
// WlinP[(ct*8+ks)*32+lane][i] = Wlin [ks*32+(lane>>4)*16+i][ct*16+(lane&15)]
// ---------------------------------------------------------------
__global__ void prep_kernel(const float* __restrict__ Wq, const float* __restrict__ Wk,
                            const float* __restrict__ Wv, const float* __restrict__ Wskip,
                            const float* __restrict__ Wlin,
                            __bf16* __restrict__ WcatP, __bf16* __restrict__ WlinP) {
    int idx = blockIdx.x * blockDim.x + threadIdx.x;
    if (idx < WCAT_CT * WCAT_KS * 32 * 16) {
        int i    = idx & 15;
        int lane = (idx >> 4) & 31;
        int ks   = (idx >> 9) & (WCAT_KS - 1);
        int ct   = idx >> 10;
        int j = ct * 16 + (lane & 15);             // fused column 0..1023
        int k = ks * 32 + (lane >> 4) * 16 + i;    // K row 0..63
        int s = j >> 8, cc = j & 255;
        const float* W = (s == 0) ? Wq : (s == 1) ? Wk : (s == 2) ? Wv : Wskip;
        WcatP[idx] = (__bf16)W[k * HC + cc];
    } else {
        int p = idx - WCAT_CT * WCAT_KS * 32 * 16;
        if (p < WLIN_CT * WLIN_KS * 32 * 16) {
            int i    = p & 15;
            int lane = (p >> 4) & 31;
            int ks   = (p >> 9) & (WLIN_KS - 1);
            int ct   = p >> 12;
            int col = ct * 16 + (lane & 15);           // 0..63
            int k   = ks * 32 + (lane >> 4) * 16 + i;  // 0..255
            WlinP[p] = (__bf16)Wlin[k * CC + col];
        }
    }
}

// ---------------------------------------------------------------
// Kernel 2: per-call init of atomic targets
// ---------------------------------------------------------------
__global__ void init_kernel(float* __restrict__ accum, float* __restrict__ denom,
                            unsigned* __restrict__ amaxU) {
    int idx = blockIdx.x * blockDim.x + threadIdx.x;
    if (idx < NN * HC) accum[idx] = 0.0f;
    if (idx < NN * HH) { denom[idx] = 0.0f; amaxU[idx] = ENC_NEG_INF; }
}

// ---------------------------------------------------------------
// Kernel 3: fused node GEMM via WMMA bf16 -> qkvs [N][1024] f32
// grid (3125, 8), block 256 (8 waves); wave w -> col tile blockIdx.y*8+w
// ---------------------------------------------------------------
__global__ __launch_bounds__(256) void gemm_nodes_kernel(const float* __restrict__ x,
                                                         const v16bf* __restrict__ WcatP,
                                                         float* __restrict__ qkvs) {
    __shared__ __bf16 Atile[16 * 64];      // 2 KB
    const int rowTile = blockIdx.x;
    const int tid = threadIdx.x;

    // stage 16x64 x-tile in LDS as bf16 (read x once for all 4 GEMMs)
    for (int i = tid; i < 16 * 64; i += 256) {
        int r = i >> 6, c = i & 63;
        Atile[i] = (__bf16)x[(size_t)(rowTile * 16 + r) * DIN + c];
    }
    __syncthreads();

    const int wave = tid >> 5;
    const int lane = tid & 31;
    const int ct   = blockIdx.y * 8 + wave;             // col tile 0..63
    const int colBase = ct * 16;
    const int row  = lane & 15;
    const int half = lane >> 4;

    v8f c = {};
#pragma unroll
    for (int ks = 0; ks < WCAT_KS; ++ks) {
        v16bf a;
        const int kbA = ks * 32 + half * 16;  // A: elems 0..15 = K kbA..+15
#pragma unroll
        for (int i = 0; i < 8; ++i) {
            a[i]     = Atile[row * 64 + kbA + i];
            a[8 + i] = Atile[row * 64 + kbA + 8 + i];
        }
        const v16bf b = WcatP[(ct * WCAT_KS + ks) * 32 + lane];  // 1x coalesced 32B load
        c = __builtin_amdgcn_wmma_f32_16x16x32_bf16(false, a, false, b, (short)0, c,
                                                    false, false);
    }
    // D: lane holds 8 rows of one column
    float* ob = qkvs + (size_t)rowTile * 16 * FUSED + colBase;
    const int col = lane & 15;
#pragma unroll
    for (int r = 0; r < 8; ++r)
        ob[(size_t)(half * 8 + r) * FUSED + col] = c[r];
}

// ---------------------------------------------------------------
// Kernel 4: alpha = q[dst]·(k[src]+e)/sqrt(C); atomic segment max
// one thread per (edge, head); e recomputed from We (11 KB, cache-resident)
// ---------------------------------------------------------------
__global__ __launch_bounds__(256) void edge_alpha_kernel(const float* __restrict__ edge_attr,
                                                         const int* __restrict__ ei,
                                                         const float* __restrict__ We,
                                                         const float* __restrict__ qkvs,
                                                         float* __restrict__ alpha,
                                                         unsigned* __restrict__ amaxU) {
    int idx = blockIdx.x * blockDim.x + threadIdx.x;
    if (idx >= EE * HH) return;
    const int h = idx & 3;
    const int e = idx >> 2;
    const int src = ei[e], dst = ei[EE + e];

    float ea[EDF];
#pragma unroll
    for (int d = 0; d < EDF; ++d) ea[d] = edge_attr[(size_t)e * EDF + d];

    const float* q   = qkvs + (size_t)dst * FUSED + h * CC;            // q section
    const float* k   = qkvs + (size_t)src * FUSED + HC + h * CC;       // k section
    const float* Wec = We + h * CC;

    float acc = 0.0f;
#pragma unroll 4
    for (int c = 0; c < CC; ++c) {
        float ec = 0.0f;
#pragma unroll
        for (int d = 0; d < EDF; ++d) ec = fmaf(ea[d], Wec[d * HC + c], ec);
        acc = fmaf(q[c], k[c] + ec, acc);
    }
    const float a = acc * 0.125f;   // 1/sqrt(64)
    alpha[idx] = a;
    atomicMax(&amaxU[(size_t)dst * HH + h], encOrdered(a));
}

// ---------------------------------------------------------------
// Kernel 5: ex = exp(alpha - amax[dst]); atomic segment sum -> denom
// ---------------------------------------------------------------
__global__ __launch_bounds__(256) void edge_exp_kernel(const int* __restrict__ ei,
                                                       const unsigned* __restrict__ amaxU,
                                                       float* __restrict__ alpha,
                                                       float* __restrict__ denom) {
    int idx = blockIdx.x * blockDim.x + threadIdx.x;
    if (idx >= EE * HH) return;
    const int h = idx & 3;
    const int e = idx >> 2;
    const int dst = ei[EE + e];
    float am = decOrdered(amaxU[(size_t)dst * HH + h]);
    if (!isfinite(am)) am = 0.0f;               // matches reference where()
    const float ex = __expf(alpha[idx] - am);
    alpha[idx] = ex;
    atomicAdd(&denom[(size_t)dst * HH + h], ex);
}

// ---------------------------------------------------------------
// Kernel 6: out[dst] += attn * (v[src] + e)   (atomic scatter)
// ---------------------------------------------------------------
__global__ __launch_bounds__(256) void edge_aggr_kernel(const float* __restrict__ edge_attr,
                                                        const int* __restrict__ ei,
                                                        const float* __restrict__ We,
                                                        const float* __restrict__ qkvs,
                                                        const float* __restrict__ alpha,
                                                        const float* __restrict__ denom,
                                                        float* __restrict__ accum) {
    int idx = blockIdx.x * blockDim.x + threadIdx.x;
    if (idx >= EE * HH) return;
    const int h = idx & 3;
    const int e = idx >> 2;
    const int src = ei[e], dst = ei[EE + e];
    const float attn = alpha[idx] / denom[(size_t)dst * HH + h];

    float ea[EDF];
#pragma unroll
    for (int d = 0; d < EDF; ++d) ea[d] = edge_attr[(size_t)e * EDF + d];

    const float* v   = qkvs + (size_t)src * FUSED + 2 * HC + h * CC;   // v section
    const float* Wec = We + h * CC;
    float* op        = accum + (size_t)dst * HC + h * CC;
#pragma unroll 4
    for (int c = 0; c < CC; ++c) {
        float ec = 0.0f;
#pragma unroll
        for (int d = 0; d < EDF; ++d) ec = fmaf(ea[d], Wec[d * HC + c], ec);
        atomicAdd(&op[c], attn * (v[c] + ec));
    }
}

// ---------------------------------------------------------------
// Kernel 7: d_out = (accum + skip) @ Wlin via WMMA bf16
// grid 3125, block 128 (4 waves -> 4 col tiles of 16)
// ---------------------------------------------------------------
__global__ __launch_bounds__(128) void final_gemm_kernel(const float* __restrict__ accum,
                                                         const float* __restrict__ qkvs,
                                                         const v16bf* __restrict__ WlinP,
                                                         float* __restrict__ out) {
    __shared__ __bf16 Htile[16 * HC];      // 8 KB
    const int rowTile = blockIdx.x;
    const int tid = threadIdx.x;

    for (int i = tid; i < 16 * HC; i += 128) {
        int r = i >> 8, cc = i & 255;
        size_t n = (size_t)rowTile * 16 + r;
        float hv = accum[n * HC + cc] + qkvs[n * FUSED + 3 * HC + cc];  // + skip
        Htile[i] = (__bf16)hv;
    }
    __syncthreads();

    const int wave = tid >> 5;                 // col tile 0..3
    const int lane = tid & 31;
    const int row  = lane & 15;
    const int half = lane >> 4;
    const int colBase = wave * 16;

    v8f c = {};
#pragma unroll
    for (int ks = 0; ks < WLIN_KS; ++ks) {
        v16bf a;
        const int kbA = ks * 32 + half * 16;
#pragma unroll
        for (int i = 0; i < 8; ++i) {
            a[i]     = Htile[row * HC + kbA + i];
            a[8 + i] = Htile[row * HC + kbA + 8 + i];
        }
        const v16bf b = WlinP[(wave * WLIN_KS + ks) * 32 + lane];  // coalesced 32B load
        c = __builtin_amdgcn_wmma_f32_16x16x32_bf16(false, a, false, b, (short)0, c,
                                                    false, false);
    }
    float* ob = out + (size_t)rowTile * 16 * CC + colBase;
    const int col = lane & 15;
#pragma unroll
    for (int r = 0; r < 8; ++r)
        ob[(size_t)(half * 8 + r) * CC + col] = c[r];
}

// ---------------------------------------------------------------
extern "C" void kernel_launch(void* const* d_in, const int* in_sizes, int n_in,
                              void* d_out, int out_size, void* d_ws, size_t ws_size,
                              hipStream_t stream) {
    const float* x         = (const float*)d_in[0];
    const float* edge_attr = (const float*)d_in[1];
    const int*   ei        = (const int*)  d_in[2];
    const float* Wq        = (const float*)d_in[3];
    const float* Wk        = (const float*)d_in[4];
    const float* Wv        = (const float*)d_in[5];
    const float* We        = (const float*)d_in[6];
    const float* Wskip     = (const float*)d_in[7];
    const float* Wlin      = (const float*)d_in[8];
    float* out = (float*)d_out;

    // ---- workspace layout (32B-aligned fragment buffers first) ----
    char* base = (char*)d_ws;
    size_t off = 0;
    __bf16* WcatP = (__bf16*)(base + off); off += (size_t)WCAT_CT * WCAT_KS * 32 * 16 * sizeof(__bf16);
    __bf16* WlinP = (__bf16*)(base + off); off += (size_t)WLIN_CT * WLIN_KS * 32 * 16 * sizeof(__bf16);
    off = (off + 255) & ~(size_t)255;
    float* qkvs   = (float*)(base + off);  off += (size_t)NN * FUSED * sizeof(float);
    float* alpha  = (float*)(base + off);  off += (size_t)EE * HH * sizeof(float);
    unsigned* amaxU = (unsigned*)(base + off); off += (size_t)NN * HH * sizeof(unsigned);
    float* denom  = (float*)(base + off);  off += (size_t)NN * HH * sizeof(float);
    float* accum  = (float*)(base + off);  off += (size_t)NN * HC * sizeof(float);
    (void)ws_size; (void)n_in; (void)in_sizes; (void)out_size;

    // 1) weight prep -> pre-packed WMMA B fragments (bf16)
    {
        int total = (WCAT_CT * WCAT_KS + WLIN_CT * WLIN_KS) * 32 * 16;
        prep_kernel<<<(total + 255) / 256, 256, 0, stream>>>(Wq, Wk, Wv, Wskip, Wlin,
                                                             WcatP, WlinP);
    }
    // 2) init atomic targets
    init_kernel<<<(NN * HC + 255) / 256, 256, 0, stream>>>(accum, denom, amaxU);

    // 3) fused node GEMM (q|k|v|skip) via WMMA
    {
        dim3 grid(NN / 16, FUSED / (16 * 8));   // (3125, 8)
        gemm_nodes_kernel<<<grid, 256, 0, stream>>>(x, (const v16bf*)WcatP, qkvs);
    }
    // 4-6) edge attention pipeline
    {
        int total = EE * HH;
        int blocks = (total + 255) / 256;
        edge_alpha_kernel<<<blocks, 256, 0, stream>>>(edge_attr, ei, We, qkvs, alpha, amaxU);
        edge_exp_kernel<<<blocks, 256, 0, stream>>>(ei, amaxU, alpha, denom);
        edge_aggr_kernel<<<blocks, 256, 0, stream>>>(edge_attr, ei, We, qkvs, alpha,
                                                     denom, accum);
    }
    // 7) trailing Linear via WMMA
    final_gemm_kernel<<<NN / 16, 128, 0, stream>>>(accum, qkvs, (const v16bf*)WlinP, out);
}
